// GNN_12936441495832
// MI455X (gfx1250) — matrix operand order repaired and checked by reading, not compiled, not run on hardware
//
#include <hip/hip_runtime.h>
#include <hip/hip_bf16.h>

typedef __attribute__((ext_vector_type(16))) _Float16 v16h;
typedef __attribute__((ext_vector_type(8)))  _Float16 v8h;
typedef __attribute__((ext_vector_type(4)))  _Float16 v4h;
typedef __attribute__((ext_vector_type(8)))  float    v8f;

#define N_NODES 100000
#define N_EDGES 600000

// =====================================================================
// WMMA GEMM:  Y[rows x dout](f16) = X[rows x din](f16) @ W^T(f32) + b
// din % 32 == 0, dout % (NT*16) == 0, rows % 16 == 0.
// block = 256 threads (8 waves). Block tile: 128 rows x NT*16 cols.
// Double-buffered LDS B tile (fragment layout), one barrier per K-step.
// B fragments register-staged, then WMMA burst (no per-WMMA ds wait).
// grid = (ceil(rows/128), dout/(NT*16))
// =====================================================================
template <int NT>
__global__ __launch_bounds__(256)
void gemm_xwT_wmma(const _Float16* __restrict__ X, const float* __restrict__ W,
                   const float* __restrict__ Bv, _Float16* __restrict__ Y,
                   int rows, int din, int dout)
{
    __shared__ _Float16 ldsB[2][NT * 32 * 16];  // 2 x NT tiles x 32 lanes x 16 halves

    const int tid  = threadIdx.x;
    const int wave = tid >> 5;
    const int lane = tid & 31;
    const int l16  = lane & 15;
    const int hi   = lane >> 4;

    const int rowBase = blockIdx.x * 128 + wave * 16;
    const int colBase = blockIdx.y * (NT * 16);

    // clamp OOB rows for loads (rows%16==0 -> wave-uniform), guard stores
    int arow = rowBase + l16;
    if (arow >= rows) arow = rows - 1;
    const _Float16* __restrict__ xrow = X + (size_t)arow * din;

    // cooperative B loader: NT*32 fragments, one per thread (tid < NT*32).
    // fragment (tile bt, lane bl): col = colBase + bt*16 + (bl&15),
    // halves h=0..15 hold K = kb + 16*(bl>>4) + h  (contiguous in W row)
    const int  bt       = tid >> 5;
    const bool isLoader = tid < NT * 32;
    const int  bcol     = colBase + (isLoader ? bt * 16 + l16 : 0);
    const int  bkoff    = hi * 16;
    const float* __restrict__ wrow = W + (size_t)bcol * din + bkoff;

    auto stage = [&](int kb, int buf) {
        if (isLoader) {
            const float4 w0 = *(const float4*)(wrow + kb + 0);
            const float4 w1 = *(const float4*)(wrow + kb + 4);
            const float4 w2 = *(const float4*)(wrow + kb + 8);
            const float4 w3 = *(const float4*)(wrow + kb + 12);
            _Float16* d = &ldsB[buf][tid * 16];
            d[0]  = (_Float16)w0.x; d[1]  = (_Float16)w0.y;
            d[2]  = (_Float16)w0.z; d[3]  = (_Float16)w0.w;
            d[4]  = (_Float16)w1.x; d[5]  = (_Float16)w1.y;
            d[6]  = (_Float16)w1.z; d[7]  = (_Float16)w1.w;
            d[8]  = (_Float16)w2.x; d[9]  = (_Float16)w2.y;
            d[10] = (_Float16)w2.z; d[11] = (_Float16)w2.w;
            d[12] = (_Float16)w3.x; d[13] = (_Float16)w3.y;
            d[14] = (_Float16)w3.z; d[15] = (_Float16)w3.w;
        }
    };

    v8f acc[NT];
#pragma unroll
    for (int t = 0; t < NT; ++t) acc[t] = (v8f){};

    const int nk = din >> 5;
    stage(0, 0);
    __syncthreads();

    for (int i = 0; i < nk; ++i) {
        const int kb = i << 5;
        if (i + 1 < nk) stage(kb + 32, (i + 1) & 1);   // uniform branch

        // prefetch A two K-tiles ahead (global_prefetch_b8)
        __builtin_prefetch(xrow + kb + 64, 0, 1);

        // A fragment: halves 0..15 -> K=kb+16*hi+h
        const v16h a = *(const v16h*)(xrow + kb + 16 * hi);

        // register-stage all B fragments, then WMMA burst
        v16h bf[NT];
#pragma unroll
        for (int t = 0; t < NT; ++t)
            bf[t] = *(const v16h*)&ldsB[i & 1][(t * 32 + lane) * 16];
#pragma unroll
        for (int t = 0; t < NT; ++t)
            acc[t] = __builtin_amdgcn_wmma_f32_16x16x32_f16(
                false, a, false, bf[t], (short)0, acc[t], false, false);

        __syncthreads();   // stage(i+1) complete; reads of buf[i&1] complete
    }

    float bb[NT];
#pragma unroll
    for (int t = 0; t < NT; ++t) bb[t] = Bv[colBase + t * 16 + l16];

    // C/D layout: lane holds N = l16, VGPR r holds M = r + 8*hi
#pragma unroll
    for (int r = 0; r < 8; ++r) {
        const int m = rowBase + r + 8 * hi;
        if (m < rows) {
            _Float16* yr = Y + (size_t)m * dout + colBase;
#pragma unroll
            for (int t = 0; t < NT; ++t)
                yr[t * 16 + l16] = (_Float16)(acc[t][r] + bb[t]);
        }
    }
}

// =====================================================================
// Layer 0 (din=3, dout=32): four linears fused, weights in LDS, f16 out
// =====================================================================
__global__ __launch_bounds__(256)
void layer0_fused(const float* __restrict__ X,
                  const float* __restrict__ Wk, const float* __restrict__ bk,
                  const float* __restrict__ Wq, const float* __restrict__ bq,
                  const float* __restrict__ Wv, const float* __restrict__ bv,
                  const float* __restrict__ Ws, const float* __restrict__ bs,
                  _Float16* __restrict__ K, _Float16* __restrict__ Q,
                  _Float16* __restrict__ V, _Float16* __restrict__ S)
{
    __shared__ float w[4 * 96];
    __shared__ float b[4 * 32];
    for (int i = threadIdx.x; i < 96; i += 256) {
        w[i] = Wk[i]; w[96 + i] = Wq[i]; w[192 + i] = Wv[i]; w[288 + i] = Ws[i];
    }
    for (int i = threadIdx.x; i < 32; i += 256) {
        b[i] = bk[i]; b[32 + i] = bq[i]; b[64 + i] = bv[i]; b[96 + i] = bs[i];
    }
    __syncthreads();

    const int n = blockIdx.x * 256 + threadIdx.x;
    if (n >= N_NODES) return;
    const float x0 = X[3 * n], x1 = X[3 * n + 1], x2 = X[3 * n + 2];

    _Float16* outs[4] = {K, Q, V, S};
#pragma unroll
    for (int m = 0; m < 4; ++m) {
        _Float16* o = outs[m] + (size_t)n * 32;
#pragma unroll
        for (int c = 0; c < 32; ++c) {
            const float* wr = &w[m * 96 + c * 3];
            o[c] = (_Float16)fmaf(x0, wr[0],
                       fmaf(x1, wr[1], fmaf(x2, wr[2], b[m * 32 + c])));
        }
    }
}

// =====================================================================
// Edge stage: eta = sigmoid(K[dst] + Q[src]); AGG[dst] += eta * V[src]
// =====================================================================
__global__ __launch_bounds__(256)
void edge_gate_scatter(const _Float16* __restrict__ Kf, const _Float16* __restrict__ Qf,
                       const _Float16* __restrict__ Vf, const long long* __restrict__ ei,
                       float* __restrict__ AGG, int C, int shift)
{
    long long t = (long long)blockIdx.x * blockDim.x + threadIdx.x;
    if (t >= ((long long)N_EDGES << shift)) return;
    const int e  = (int)(t >> shift);
    const int c4 = ((int)t & ((1 << shift) - 1)) << 2;
    const long long src = ei[2 * e + 0];
    const long long dst = ei[2 * e + 1];

    const v4h kk = *(const v4h*)(Kf + dst * (long long)C + c4);
    const v4h qq = *(const v4h*)(Qf + src * (long long)C + c4);
    const v4h vv = *(const v4h*)(Vf + src * (long long)C + c4);

    float4 m;
    m.x = (float)vv[0] / (1.0f + __expf(-((float)kk[0] + (float)qq[0])));
    m.y = (float)vv[1] / (1.0f + __expf(-((float)kk[1] + (float)qq[1])));
    m.z = (float)vv[2] / (1.0f + __expf(-((float)kk[2] + (float)qq[2])));
    m.w = (float)vv[3] / (1.0f + __expf(-((float)kk[3] + (float)qq[3])));

    float* d = AGG + dst * (long long)C + c4;
    atomicAdd(d + 0, m.x);
    atomicAdd(d + 1, m.y);
    atomicAdd(d + 2, m.z);
    atomicAdd(d + 3, m.w);
}

// =====================================================================
// Node stage: H = relu(AGG + S)  (f32 + f16 -> f16 activations)
// =====================================================================
__global__ __launch_bounds__(256)
void node_finish(const float* __restrict__ AGG, const _Float16* __restrict__ S,
                 _Float16* __restrict__ H, long long n)
{
    long long i = ((long long)blockIdx.x * blockDim.x + threadIdx.x) * 4;
    if (i >= n) return;
    const float4 a = *(const float4*)(AGG + i);
    const v4h    s = *(const v4h*)(S + i);
    v4h o;
    o[0] = (_Float16)fmaxf(a.x + (float)s[0], 0.0f);
    o[1] = (_Float16)fmaxf(a.y + (float)s[1], 0.0f);
    o[2] = (_Float16)fmaxf(a.z + (float)s[2], 0.0f);
    o[3] = (_Float16)fmaxf(a.w + (float)s[3], 0.0f);
    *(v4h*)(H + i) = o;
}

// =====================================================================
// Output projection: out[n,0..2] = h[n,:512] . out_w[o,:512] + out_b[o]
// =====================================================================
__global__ __launch_bounds__(256)
void out_proj(const _Float16* __restrict__ H, const float* __restrict__ Wout,
              const float* __restrict__ bout, float* __restrict__ OUT)
{
    __shared__ float w[3 * 512];
    for (int i = threadIdx.x; i < 3 * 512; i += blockDim.x) w[i] = Wout[i];
    __syncthreads();

    const int nidx = blockIdx.x * blockDim.x + threadIdx.x;
    if (nidx >= N_NODES) return;
    const _Float16* h = H + (size_t)nidx * 512;
    float o0 = bout[0], o1 = bout[1], o2 = bout[2];
#pragma unroll 2
    for (int k = 0; k < 512; k += 4) {
        const v4h hv4 = *(const v4h*)(h + k);
#pragma unroll
        for (int j = 0; j < 4; ++j) {
            const float hv = (float)hv4[j];
            o0 = fmaf(hv, w[k + j],        o0);
            o1 = fmaf(hv, w[512 + k + j],  o1);
            o2 = fmaf(hv, w[1024 + k + j], o2);
        }
    }
    OUT[(size_t)nidx * 3 + 0] = o0;
    OUT[(size_t)nidx * 3 + 1] = o1;
    OUT[(size_t)nidx * 3 + 2] = o2;
}

// =====================================================================
extern "C" void kernel_launch(void* const* d_in, const int* in_sizes, int n_in,
                              void* d_out, int out_size, void* d_ws, size_t ws_size,
                              hipStream_t stream)
{
    (void)in_sizes; (void)n_in; (void)out_size; (void)ws_size;

    static const int DIMS[11][2] = {
        {3,32},{32,64},{64,128},{128,128},{128,128},{128,256},
        {256,256},{256,256},{256,512},{512,512},{512,512}
    };

    const float*     x  = (const float*)d_in[0];
    const long long* ei = (const long long*)d_in[1];   // int64 edge_index [E,2]

    // workspace layout: AGG f32 slab, then 5 f16 slabs (H, K, Q, V, S)
    const size_t slabE = (size_t)N_NODES * 512;
    float*     AGG = (float*)d_ws;
    _Float16*  f16base = (_Float16*)((char*)d_ws + slabE * sizeof(float));
    _Float16*  Hb = f16base + 0 * slabE;
    _Float16*  Kb = f16base + 1 * slabE;
    _Float16*  Qb = f16base + 2 * slabE;
    _Float16*  Vb = f16base + 3 * slabE;
    _Float16*  Sb = f16base + 4 * slabE;

    for (int L = 0; L < 11; ++L) {
        const int din  = DIMS[L][0];
        const int dout = DIMS[L][1];
        const float* Wk = (const float*)d_in[2 + L * 8 + 0];
        const float* bk = (const float*)d_in[2 + L * 8 + 1];
        const float* Wq = (const float*)d_in[2 + L * 8 + 2];
        const float* bq = (const float*)d_in[2 + L * 8 + 3];
        const float* Wv = (const float*)d_in[2 + L * 8 + 4];
        const float* bv = (const float*)d_in[2 + L * 8 + 5];
        const float* Ws = (const float*)d_in[2 + L * 8 + 6];
        const float* bs = (const float*)d_in[2 + L * 8 + 7];

        if (L == 0) {
            layer0_fused<<<(N_NODES + 255) / 256, 256, 0, stream>>>(
                x, Wk, bk, Wq, bq, Wv, bv, Ws, bs, Kb, Qb, Vb, Sb);
        } else if (dout % 128 == 0) {
            dim3 gg((N_NODES + 127) / 128, dout / 128);
            gemm_xwT_wmma<8><<<gg, 256, 0, stream>>>(Hb, Wk, bk, Kb, N_NODES, din, dout);
            gemm_xwT_wmma<8><<<gg, 256, 0, stream>>>(Hb, Wq, bq, Qb, N_NODES, din, dout);
            gemm_xwT_wmma<8><<<gg, 256, 0, stream>>>(Hb, Wv, bv, Vb, N_NODES, din, dout);
            gemm_xwT_wmma<8><<<gg, 256, 0, stream>>>(Hb, Ws, bs, Sb, N_NODES, din, dout);
        } else {
            dim3 gg((N_NODES + 127) / 128, dout / 64);
            gemm_xwT_wmma<4><<<gg, 256, 0, stream>>>(Hb, Wk, bk, Kb, N_NODES, din, dout);
            gemm_xwT_wmma<4><<<gg, 256, 0, stream>>>(Hb, Wq, bq, Qb, N_NODES, din, dout);
            gemm_xwT_wmma<4><<<gg, 256, 0, stream>>>(Hb, Wv, bv, Vb, N_NODES, din, dout);
            gemm_xwT_wmma<4><<<gg, 256, 0, stream>>>(Hb, Ws, bs, Sb, N_NODES, din, dout);
        }

        hipMemsetAsync(AGG, 0, (size_t)N_NODES * dout * sizeof(float), stream);

        int shift = 0;
        while ((4 << shift) < dout) ++shift;   // log2(dout/4)
        long long work = (long long)N_EDGES << shift;
        edge_gate_scatter<<<(unsigned)((work + 255) / 256), 256, 0, stream>>>(
            Kb, Qb, Vb, ei, AGG, dout, shift);

        long long n = (long long)N_NODES * dout;
        node_finish<<<(unsigned)((n / 4 + 255) / 256), 256, 0, stream>>>(
            AGG, Sb, Hb, n);
    }

    out_proj<<<(N_NODES + 255) / 256, 256, 0, stream>>>(
        Hb, (const float*)d_in[90], (const float*)d_in[91], (float*)d_out);
}
